// CGBatched_13408887898250
// MI455X (gfx1250) — matrix operand order repaired and checked by reading, not compiled, not run on hardware
//
#include <hip/hip_runtime.h>
#include <hip/hip_bf16.h>

// ---- CDNA5 WMMA vector types -----------------------------------------------
typedef __attribute__((ext_vector_type(16))) __bf16 v16bf;
typedef __attribute__((ext_vector_type(8)))  __bf16 v8bf;
typedef __attribute__((ext_vector_type(4)))  __bf16 v4bf;
typedef __attribute__((ext_vector_type(8)))  float  v8f;

#define CG_N       1024
#define CG_THREADS 512
#define CG_NWAVES  (CG_THREADS / 32)
#define CG_ROWS_PER_WAVE 64            // 4 x 16-row WMMA blocks per wave

// ---------------------------------------------------------------------------
// Kernel 1: S = bf16(A - I), streamed once.  4 elements / thread.
// ---------------------------------------------------------------------------
__global__ void cg_convert_S(const float* __restrict__ A,
                             __bf16* __restrict__ S,
                             size_t total4) {
    size_t t = (size_t)blockIdx.x * blockDim.x + threadIdx.x;
    if (t >= total4) return;
    size_t e = t * 4;                               // first of 4 contiguous elems
    const float4 v = *(const float4*)(A + e);
    size_t within = e & (size_t)(CG_N * CG_N - 1);  // index inside one N*N matrix
    int r = (int)(within >> 10);                    // row
    int c = (int)(within & (CG_N - 1));             // col of element 0
    float f[4] = {v.x, v.y, v.z, v.w};
    int d = r - c;                                  // diagonal falls at f[d] if 0..3
    if (d >= 0 && d < 4) f[d] -= 1.0f;
    v4bf o;
    o[0] = (__bf16)f[0]; o[1] = (__bf16)f[1];
    o[2] = (__bf16)f[2]; o[3] = (__bf16)f[3];
    *(v4bf*)(S + e) = o;
}

// ---------------------------------------------------------------------------
// Wave-cooperative y = S * v  (S: N x N bf16 row-major, v: LDS bf16, y: LDS f32)
//
// k0 is the OUTER loop (unroll 1) so only ONE B-fragment lives in registers at
// a time; the 4 row-blocks per wave are the inner, fully-unrolled loop with 4
// persistent f32 accumulators.  Each wave owns 64 contiguous rows, so the four
// A-tile loads share one base pointer + immediate offsets (j * 32768 B).
//
// A-fragment lane layout (16-bit A 16x32, contiguous k order matching B):
//   lane l, row = rowBase + (l & 15), half = l >> 4
//   elems 0..15 = S[row][k0 + 16*half .. +15]
// B-fragment (32x16, v replicated over all 16 cols):
//   lane l: elems 0..15 = v[k0 + 16*half .. +15]   (one contiguous 32B LDS read)
// C/D: VGPR i of lane-half h holds row rowBase + i + 8*h.
// ---------------------------------------------------------------------------
__device__ __forceinline__ void matvec_bf16(const __bf16* __restrict__ S,
                                            const __bf16* __restrict__ vb,
                                            float* __restrict__ y,
                                            int wave, int lane) {
    const int half = lane >> 4;
    const int mrow = lane & 15;
    const int rowBase = wave * CG_ROWS_PER_WAVE;          // 64 contiguous rows
    const __bf16* base = S + (size_t)(rowBase + mrow) * CG_N + half * 16;

    v8f acc0 = {0.f, 0.f, 0.f, 0.f, 0.f, 0.f, 0.f, 0.f};
    v8f acc1 = acc0, acc2 = acc0, acc3 = acc0;

#pragma unroll 1
    for (int k0 = 0; k0 < CG_N; k0 += 32) {
        const v16bf b = *(const v16bf*)(vb + k0 + 16 * half);
        const __bf16* rp = base + k0;

#define CG_TILE(J, ACC)                                                        \
        {                                                                      \
            v8bf lo = *(const v8bf*)(rp + (size_t)(J) * 16 * CG_N);            \
            v8bf hi = *(const v8bf*)(rp + (size_t)(J) * 16 * CG_N + 8);        \
            v16bf a = __builtin_shufflevector(lo, hi, 0, 1, 2, 3, 4, 5, 6, 7,  \
                                              8, 9, 10, 11, 12, 13, 14, 15);  \
            ACC = __builtin_amdgcn_wmma_f32_16x16x32_bf16(                     \
                false, a, false, b, (short)0, ACC, false, false);              \
        }
        CG_TILE(0, acc0)
        CG_TILE(1, acc1)
        CG_TILE(2, acc2)
        CG_TILE(3, acc3)
#undef CG_TILE
    }

    if (mrow == 0) {                    // lane 0 -> rows 0..7, lane 16 -> 8..15
        float* yp = y + rowBase + half * 8;
#pragma unroll
        for (int i = 0; i < 8; ++i) {
            yp[i]      = acc0[i];
            yp[16 + i] = acc1[i];
            yp[32 + i] = acc2[i];
            yp[48 + i] = acc3[i];
        }
    }
}

// fp32 fallback mat-vec (full A, one row per wave-iteration, lane-strided K)
__device__ __forceinline__ void matvec_f32(const float* __restrict__ A,
                                           const float* __restrict__ p,
                                           float* __restrict__ y,
                                           int wave, int lane) {
    for (int row = wave; row < CG_N; row += CG_NWAVES) {
        const float* rp = A + (size_t)row * CG_N;
        float s = 0.f;
        for (int k = lane; k < CG_N; k += 32) s += rp[k] * p[k];
        for (int off = 16; off > 0; off >>= 1) s += __shfl_down(s, off, 32);
        if (lane == 0) y[row] = s;
    }
}

// deterministic block-wide dot over N=1024 (2 elems / thread, LDS tree)
__device__ __forceinline__ float block_dot(const float* a, const float* b,
                                           float* red, int tid) {
    red[tid] = a[tid] * b[tid] + a[tid + CG_THREADS] * b[tid + CG_THREADS];
    __syncthreads();
    for (int off = CG_THREADS / 2; off > 0; off >>= 1) {
        if (tid < off) red[tid] += red[tid + off];
        __syncthreads();
    }
    float r = red[0];
    __syncthreads();
    return r;
}

// ---------------------------------------------------------------------------
// Kernel 2: full CG for one batch per workgroup (vectors live in LDS).
// ---------------------------------------------------------------------------
template <bool USE_BF16>
__global__ __launch_bounds__(CG_THREADS)
void cg_solve(const float* __restrict__ u, const float* __restrict__ bvec,
              const float* __restrict__ A, const __bf16* __restrict__ S,
              const int* __restrict__ maxiter_p, float* __restrict__ out) {
    __shared__ float xs[CG_N], rs[CG_N], ps[CG_N], Aps[CG_N];
    __shared__ __align__(32) __bf16 vb[CG_N];
    __shared__ float red[CG_THREADS];

    const int tid = threadIdx.x;
    const int lane = tid & 31, wave = tid >> 5;
    const int batch = blockIdx.x;
    const int i0 = tid, i1 = tid + CG_THREADS;

    const float*  ub = u    + (size_t)batch * CG_N;
    const float*  bb = bvec + (size_t)batch * CG_N;
    const float*  Ab = A    + (size_t)batch * CG_N * CG_N;
    const __bf16* Sb = USE_BF16 ? (S + (size_t)batch * CG_N * CG_N) : nullptr;

    const float b0 = bb[i0], b1 = bb[i1];
    xs[i0] = ub[i0]; xs[i1] = ub[i1];
    if (USE_BF16) { vb[i0] = (__bf16)xs[i0]; vb[i1] = (__bf16)xs[i1]; }
    __syncthreads();

    // r0 = b - A x0   (A x = S x + x in the bf16 path)
    if constexpr (USE_BF16) matvec_bf16(Sb, vb, Aps, wave, lane);
    else                    matvec_f32(Ab, xs, Aps, wave, lane);
    __syncthreads();
    {
        float ax0 = USE_BF16 ? (Aps[i0] + xs[i0]) : Aps[i0];
        float ax1 = USE_BF16 ? (Aps[i1] + xs[i1]) : Aps[i1];
        rs[i0] = b0 - ax0; rs[i1] = b1 - ax1;
        ps[i0] = rs[i0];   ps[i1] = rs[i1];
        if (USE_BF16) { vb[i0] = (__bf16)ps[i0]; vb[i1] = (__bf16)ps[i1]; }
    }
    float rr = block_dot(rs, rs, red, tid);   // barriers publish ps/vb too

    const int maxiter = maxiter_p[0];
    for (int it = 0; it < maxiter; ++it) {
        if constexpr (USE_BF16) matvec_bf16(Sb, vb, Aps, wave, lane);
        else                    matvec_f32(Ab, ps, Aps, wave, lane);
        __syncthreads();
        if (USE_BF16) { Aps[i0] += ps[i0]; Aps[i1] += ps[i1]; }  // Ap = Sp + p

        float pAp   = block_dot(ps, Aps, red, tid);
        float alpha = rr / pAp;
        xs[i0] += alpha * ps[i0];  xs[i1] += alpha * ps[i1];
        rs[i0] -= alpha * Aps[i0]; rs[i1] -= alpha * Aps[i1];

        float rrn  = block_dot(rs, rs, red, tid);
        float beta = rrn / rr;
        rr = rrn;
        ps[i0] = rs[i0] + beta * ps[i0];
        ps[i1] = rs[i1] + beta * ps[i1];
        if (USE_BF16) { vb[i0] = (__bf16)ps[i0]; vb[i1] = (__bf16)ps[i1]; }
        __syncthreads();                      // ps/vb visible for next matvec
    }

    out[(size_t)batch * CG_N + i0] = xs[i0];
    out[(size_t)batch * CG_N + i1] = xs[i1];
}

// ---------------------------------------------------------------------------
extern "C" void kernel_launch(void* const* d_in, const int* in_sizes, int n_in,
                              void* d_out, int out_size, void* d_ws, size_t ws_size,
                              hipStream_t stream) {
    const float* u  = (const float*)d_in[0];   // [B,32,32] == x0
    const float* b  = (const float*)d_in[1];   // [B,N,1]
    const float* A  = (const float*)d_in[2];   // [B,N,N]
    const int*   mi = (const int*)d_in[3];     // maxiter scalar
    float* out = (float*)d_out;

    const int    Bn     = in_sizes[0] / CG_N;                 // 256
    const size_t totalS = (size_t)Bn * CG_N * CG_N;           // elements of A
    const size_t needed = totalS * sizeof(__bf16);            // 512 MB

    if (ws_size >= needed) {
        __bf16* S = (__bf16*)d_ws;
        const size_t total4  = totalS / 4;
        const int    threads = 256;
        const int    blocks  = (int)((total4 + threads - 1) / threads);
        cg_convert_S<<<blocks, threads, 0, stream>>>(A, S, total4);
        cg_solve<true><<<Bn, CG_THREADS, 0, stream>>>(u, b, A, S, mi, out);
    } else {
        cg_solve<false><<<Bn, CG_THREADS, 0, stream>>>(u, b, A, nullptr, mi, out);
    }
}
